// CustomMultiHeadSelfAttention_25417616458322
// MI455X (gfx1250) — matrix operand order repaired and checked by reading, not compiled, not run on hardware
//
#include <hip/hip_runtime.h>

typedef __attribute__((ext_vector_type(16))) __bf16        v16bf;
typedef __attribute__((ext_vector_type(8)))  float         v8f;
typedef __attribute__((ext_vector_type(4)))  unsigned int  v4u;
typedef __attribute__((ext_vector_type(8)))  int           v8i;
typedef __attribute__((ext_vector_type(4)))  int           v4i;

#if __has_builtin(__builtin_amdgcn_tensor_load_to_lds) && __has_builtin(__builtin_amdgcn_s_wait_tensorcnt)
#define USE_TDM 1
#else
#define USE_TDM 0
#endif

#if __has_builtin(__builtin_amdgcn_permlane16)
#define USE_PERMLANE 1
#else
#define USE_PERMLANE 0
#endif

// ---------- helpers ----------

__device__ __forceinline__ unsigned short f2bf(float f) {
    unsigned int u = __float_as_uint(f);
    unsigned int r = u + 0x7FFFu + ((u >> 16) & 1u);   // round-to-nearest-even
    return (unsigned short)(r >> 16);
}

__device__ __forceinline__ v8f wmma_bf16(v16bf a, v16bf b, v8f c) {
    return __builtin_amdgcn_wmma_f32_16x16x32_bf16(
        false, a, false, b, (short)0, c, false, false);
}

// XOR-butterfly exchange within 16-lane halves: v_permlane16_b32 (VALU) if available.
__device__ __forceinline__ float xor16_f(float v, int k, unsigned sel0, unsigned sel1) {
#if USE_PERMLANE
    (void)k;
    unsigned u = __float_as_uint(v);
    unsigned r = __builtin_amdgcn_permlane16(u, u, sel0, sel1, false, false);
    return __uint_as_float(r);
#else
    (void)sel0; (void)sel1;
    return __shfl_xor(v, k, 16);
#endif
}

__device__ __forceinline__ float max16(float v) {
    v = fmaxf(v, xor16_f(v, 1, 0x67452301u, 0xEFCDAB89u));
    v = fmaxf(v, xor16_f(v, 2, 0x54761032u, 0xDCFE98BAu));
    v = fmaxf(v, xor16_f(v, 4, 0x32107654u, 0xBA98FEDCu));
    v = fmaxf(v, xor16_f(v, 8, 0xFEDCBA98u, 0x76543210u));
    return v;
}
__device__ __forceinline__ float sum16(float v) {
    v += xor16_f(v, 1, 0x67452301u, 0xEFCDAB89u);
    v += xor16_f(v, 2, 0x54761032u, 0xDCFE98BAu);
    v += xor16_f(v, 4, 0x32107654u, 0xBA98FEDCu);
    v += xor16_f(v, 8, 0xFEDCBA98u, 0x76543210u);
    return v;
}

// A-fragment (16x32 bf16, M x K). prow = pointer to this lane's row (M = lane&15).
__device__ __forceinline__ v16bf a_frag_from_rowptr(const unsigned short* prow, int half) {
    union { v16bf v; unsigned int u[8]; } r;
#pragma unroll
    for (int j = 0; j < 8; ++j) {
        int kk = (j < 4) ? (half * 8 + 2 * j) : (16 + half * 8 + 2 * (j - 4));
        r.u[j] = *(const unsigned int*)(prow + kk);
    }
    return r.v;
}

// B-fragment (32x16 bf16, K x N) from K-contiguous (transposed) storage.
__device__ __forceinline__ v16bf b_frag_from_colptr(const unsigned short* pcol, int half) {
    union { v16bf v; unsigned int u[8]; } r;
    const unsigned short* p = pcol;
#pragma unroll
    for (int j = 0; j < 8; ++j) r.u[j] = *(const unsigned int*)(p + ((j < 4) ? (half * 8 + 2 * j) : (16 + half * 8 + 2 * (j - 4))));
    return r.v;
}

// TDM: async DMA of a 2D bf16 tile (width 64 elems, `rows` lines, row stride
// `stride_elems`) from global into LDS at lds_byte_addr with +16B padding every
// 128B (-> LDS row stride 72 bf16, matching the bank-padded layouts below).
// This toolchain's builtin is the 6-arg form: (g0 v4u, g1 v8i, g2 v4i, g3 v4i, g4 v8i, cpol).
__device__ __forceinline__ void tdm_load_64xN_bf16(unsigned lds_byte_addr, const void* gaddr,
                                                   unsigned rows, unsigned stride_elems) {
#if USE_TDM
    unsigned long long ga = (unsigned long long)(uintptr_t)gaddr;
    v4u g0;
    g0.x = 1u;                                                   // count=1 (valid user D#)
    g0.y = lds_byte_addr;                                        // lds_addr
    g0.z = (unsigned)ga;                                         // global_addr[31:0]
    g0.w = ((unsigned)(ga >> 32) & 0x01FFFFFFu) | (2u << 30);    // global_addr[56:32], type=2
    v8i g1;
    g1[0] = (int)((1u << 16) | (1u << 20) | (4u << 22) | (3u << 25));
    //            data_size=2B  pad_enable  pad_intvl=32dw  pad_amt=4dw
    g1[1] = (int)(64u << 16);          // tensor_dim0 = 64 (bits 63:48)
    g1[2] = (int)(rows << 16);         // tensor_dim1 (bits 95:80)
    g1[3] = (int)(64u << 16);          // tile_dim0 = 64 (bits 127:112)
    g1[4] = (int)rows;                 // tile_dim1 (bits 143:128)
    g1[5] = (int)stride_elems;         // tensor_dim0_stride[31:0] (bits 191:160)
    g1[6] = 0; g1[7] = 0;
    v4i z4 = {0, 0, 0, 0};
    v8i z8 = {0, 0, 0, 0, 0, 0, 0, 0};
    __builtin_amdgcn_tensor_load_to_lds(g0, g1, z4, z4, z8, 0);
#else
    (void)lds_byte_addr; (void)gaddr; (void)rows; (void)stride_elems;
#endif
}

// ---------- fp32 -> bf16 conversion ----------

__global__ void cvt_f32_bf16(const float2* __restrict__ in, unsigned int* __restrict__ out, int n2) {
    for (int i = blockIdx.x * blockDim.x + threadIdx.x; i < n2; i += gridDim.x * blockDim.x) {
        float2 v = in[i];
        out[i] = (unsigned int)f2bf(v.x) | ((unsigned int)f2bf(v.y) << 16);
    }
}

// ---------- bf16 GEMM: C[M,N] = A[M,K] * B[K,N] + bias ----------
// Block tile 64x128, 8 waves (2x4), per-wave 32x32 (2x2 WMMA tiles), K staged 64 at a time.
// A tile DMA'd by the TDM (row-major + HW padding); B tile thread-staged transposed.

template <int MODE>
__global__ __launch_bounds__(256) void gemm_bf16_kernel(
    const unsigned short* __restrict__ A, const unsigned short* __restrict__ B,
    const float* __restrict__ bias,
    unsigned short* __restrict__ oq, unsigned short* __restrict__ ok_, unsigned short* __restrict__ ov,
    float* __restrict__ of, int M, int N, int K)
{
    __shared__ __attribute__((aligned(16))) unsigned short Alds[64 * 72];   // [m][k], stride 72
    __shared__ __attribute__((aligned(16))) unsigned short Blds[128 * 72];  // [n][k] (transposed)

    const int tid = threadIdx.x;
    const int lane = tid & 31, wave = tid >> 5;
    const int wm = wave >> 2, wn = wave & 3;
    const int half = lane >> 4, l15 = lane & 15;
    const int m0 = blockIdx.y * 64;
    const int n0 = blockIdx.x * 128;
    const unsigned alds_off = (unsigned)(uintptr_t)(&Alds[0]);   // LDS aperture: low 32 bits = LDS byte offset

    v8f acc[2][2] = {};

    for (int k0 = 0; k0 < K; k0 += 64) {
#if USE_TDM
        if (wave == 0)
            tdm_load_64xN_bf16(alds_off, A + (size_t)m0 * K + k0, 64, (unsigned)K);
#else
        // stage A: 64x64 bf16 = 2048 dwords, 8 per thread
#pragma unroll
        for (int i = 0; i < 8; ++i) {
            int p = tid + 256 * i;
            int m = p >> 5, kc = (p & 31) * 2;
            *(unsigned int*)(&Alds[m * 72 + kc]) =
                *(const unsigned int*)(A + (size_t)(m0 + m) * K + k0 + kc);
        }
#endif
        // stage B transposed: 64x128 bf16 = 4096 dwords-worth, 16 dword loads per thread
#pragma unroll
        for (int i = 0; i < 16; ++i) {
            int p = tid + 256 * i;
            int k = p >> 6, nn = (p & 63) * 2;
            unsigned int w = *(const unsigned int*)(B + (size_t)(k0 + k) * N + n0 + nn);
            Blds[nn * 72 + k]       = (unsigned short)(w & 0xffffu);
            Blds[(nn + 1) * 72 + k] = (unsigned short)(w >> 16);
        }
#if USE_TDM
        if (wave == 0) __builtin_amdgcn_s_wait_tensorcnt(0);
#endif
        __syncthreads();

        v16bf a0l = a_frag_from_rowptr(&Alds[(wm * 32 + l15) * 72], half);
        v16bf a0h = a_frag_from_rowptr(&Alds[(wm * 32 + l15) * 72 + 32], half);
        v16bf a1l = a_frag_from_rowptr(&Alds[(wm * 32 + 16 + l15) * 72], half);
        v16bf a1h = a_frag_from_rowptr(&Alds[(wm * 32 + 16 + l15) * 72 + 32], half);
        v16bf b0l = b_frag_from_colptr(&Blds[(wn * 32 + l15) * 72], half);
        v16bf b0h = b_frag_from_colptr(&Blds[(wn * 32 + l15) * 72 + 32], half);
        v16bf b1l = b_frag_from_colptr(&Blds[(wn * 32 + 16 + l15) * 72], half);
        v16bf b1h = b_frag_from_colptr(&Blds[(wn * 32 + 16 + l15) * 72 + 32], half);

        acc[0][0] = wmma_bf16(a0l, b0l, acc[0][0]);
        acc[0][0] = wmma_bf16(a0h, b0h, acc[0][0]);
        acc[0][1] = wmma_bf16(a0l, b1l, acc[0][1]);
        acc[0][1] = wmma_bf16(a0h, b1h, acc[0][1]);
        acc[1][0] = wmma_bf16(a1l, b0l, acc[1][0]);
        acc[1][0] = wmma_bf16(a1h, b0h, acc[1][0]);
        acc[1][1] = wmma_bf16(a1l, b1l, acc[1][1]);
        acc[1][1] = wmma_bf16(a1h, b1h, acc[1][1]);
        __syncthreads();
    }

#pragma unroll
    for (int tm = 0; tm < 2; ++tm)
#pragma unroll
        for (int tn = 0; tn < 2; ++tn)
#pragma unroll
            for (int r = 0; r < 8; ++r) {
                int gm = m0 + wm * 32 + tm * 16 + r + 8 * half;
                int gn = n0 + wn * 32 + tn * 16 + l15;
                float v = acc[tm][tn][r] + bias[gn];
                if (MODE == 0) {
                    int which = gn >> 10, cl = gn & 1023;
                    if (which == 0)      oq [(size_t)gm * 1024 + cl] = f2bf(v * 0.125f); // fold 1/sqrt(64)
                    else if (which == 1) ok_[(size_t)gm * 1024 + cl] = f2bf(v);
                    else                 ov [(size_t)gm * 1024 + cl] = f2bf(v);
                } else {
                    of[(size_t)gm * N + gn] = v;
                }
            }
}

// ---------- fused flash attention (bf16 WMMA, fp32 softmax stats) ----------
// grid: (heads=16, qblocks=16). 8 waves x 16 query rows = 128 rows per block.
// 64 keys per iteration: 16 WMMAs between barriers; K tile via TDM, V transposed by threads.

__global__ __launch_bounds__(256) void attn_fwd_kernel(
    const unsigned short* __restrict__ Q, const unsigned short* __restrict__ Kb,
    const unsigned short* __restrict__ Vb, unsigned short* __restrict__ O)
{
    __shared__ __attribute__((aligned(16))) unsigned short Klds[64 * 72];      // [key][d], stride 72
    __shared__ __attribute__((aligned(16))) unsigned short Vlds[64 * 72];      // [d][key] (transposed)
    __shared__ __attribute__((aligned(16))) unsigned short Plds[8 * 16 * 72];  // per-wave [row][key]

    const int tid = threadIdx.x, lane = tid & 31, wave = tid >> 5;
    const int half = lane >> 4, l15 = lane & 15;
    const int head = blockIdx.x;
    const int qb0 = blockIdx.y * 128;
    const unsigned klds_off = (unsigned)(uintptr_t)(&Klds[0]);

    const int qrow = qb0 + wave * 16 + l15;
    const unsigned short* qp = Q + (size_t)qrow * 1024 + head * 64;
    v16bf aq0 = a_frag_from_rowptr(qp, half);        // d 0..31 (pre-scaled by 1/sqrt(D))
    v16bf aq1 = a_frag_from_rowptr(qp + 32, half);   // d 32..63

    float mrow[8], lrow[8];
    v8f acc[4] = {};
#pragma unroll
    for (int r = 0; r < 8; ++r) { mrow[r] = -1e30f; lrow[r] = 0.0f; }

    unsigned short* pw = &Plds[wave * 16 * 72];

    for (int j0 = 0; j0 < 2048; j0 += 64) {
        __syncthreads();   // previous iteration's K/V fragment reads done
#if USE_TDM
        if (wave == 0)
            tdm_load_64xN_bf16(klds_off, Kb + (size_t)j0 * 1024 + head * 64, 64, 1024u);
#else
        // stage K row-major: 64x64 bf16, 8 dwords/thread
#pragma unroll
        for (int i = 0; i < 8; ++i) {
            int p = tid + 256 * i;
            int r = p >> 5, cu = (p & 31) * 2;
            *(unsigned int*)(&Klds[r * 72 + cu]) =
                *(const unsigned int*)(Kb + (size_t)(j0 + r) * 1024 + head * 64 + cu);
        }
#endif
        // stage V transposed: 64x64 bf16, 8 dwords/thread
#pragma unroll
        for (int i = 0; i < 8; ++i) {
            int p = tid + 256 * i;
            int r = p >> 5, cu = (p & 31) * 2;
            unsigned int w = *(const unsigned int*)(Vb + (size_t)(j0 + r) * 1024 + head * 64 + cu);
            Vlds[cu * 72 + r]       = (unsigned short)(w & 0xffffu);
            Vlds[(cu + 1) * 72 + r] = (unsigned short)(w >> 16);
        }
        if (j0 + 64 < 2048) {
            __builtin_prefetch(Kb + (size_t)(j0 + 64 + lane) * 1024 + head * 64, 0, 1);
            __builtin_prefetch(Vb + (size_t)(j0 + 64 + lane) * 1024 + head * 64, 0, 1);
        }
#if USE_TDM
        if (wave == 0) __builtin_amdgcn_s_wait_tensorcnt(0);
#endif
        __syncthreads();

        // scores: S[16 x 64] = Q[16 x 64] * K^T, four 16x16 tiles, 2 WMMAs each
        v8f s[4];
#pragma unroll
        for (int t = 0; t < 4; ++t) {
            v8f z = {};
            v16bf bk0 = b_frag_from_colptr(&Klds[(t * 16 + l15) * 72], half);       // d 0-31
            v16bf bk1 = b_frag_from_colptr(&Klds[(t * 16 + l15) * 72 + 32], half);  // d 32-63
            z = wmma_bf16(aq0, bk0, z);
            z = wmma_bf16(aq1, bk1, z);
            s[t] = z;
        }

        // online softmax (VGPR r <-> row r + 8*half; rows live across 16-lane halves)
#pragma unroll
        for (int r = 0; r < 8; ++r) {
            float rowmax = fmaxf(fmaxf(s[0][r], s[1][r]), fmaxf(s[2][r], s[3][r]));
            rowmax = max16(rowmax);
            float mnew = fmaxf(mrow[r], rowmax);
            float f = __expf(mrow[r] - mnew);
            float p0 = __expf(s[0][r] - mnew);
            float p1 = __expf(s[1][r] - mnew);
            float p2 = __expf(s[2][r] - mnew);
            float p3 = __expf(s[3][r] - mnew);
            float rs = sum16((p0 + p1) + (p2 + p3));
            lrow[r] = lrow[r] * f + rs;
            mrow[r] = mnew;
            acc[0][r] *= f; acc[1][r] *= f; acc[2][r] *= f; acc[3][r] *= f;
            int prow = r + 8 * half;
            pw[prow * 72 + l15]      = f2bf(p0);
            pw[prow * 72 + 16 + l15] = f2bf(p1);
            pw[prow * 72 + 32 + l15] = f2bf(p2);
            pw[prow * 72 + 48 + l15] = f2bf(p3);
        }

        // O += P[16 x 64] * V[64 x 64]  (same-wave LDS ops are in-order)
        v16bf pa0 = a_frag_from_rowptr(&pw[l15 * 72], half);        // keys 0-31
        v16bf pa1 = a_frag_from_rowptr(&pw[l15 * 72 + 32], half);   // keys 32-63
#pragma unroll
        for (int c = 0; c < 4; ++c) {
            v16bf bv0 = b_frag_from_colptr(&Vlds[(c * 16 + l15) * 72], half);
            v16bf bv1 = b_frag_from_colptr(&Vlds[(c * 16 + l15) * 72 + 32], half);
            acc[c] = wmma_bf16(pa0, bv0, acc[c]);
            acc[c] = wmma_bf16(pa1, bv1, acc[c]);
        }
    }

    // normalize and write context (bf16)
#pragma unroll
    for (int c = 0; c < 4; ++c)
#pragma unroll
        for (int r = 0; r < 8; ++r) {
            float o = acc[c][r] / lrow[r];
            int row = qb0 + wave * 16 + r + 8 * half;
            int col = head * 64 + c * 16 + l15;
            O[(size_t)row * 1024 + col] = f2bf(o);
        }
}

// ---------- launch ----------

extern "C" void kernel_launch(void* const* d_in, const int* in_sizes, int n_in,
                              void* d_out, int out_size, void* d_ws, size_t ws_size,
                              hipStream_t stream) {
    (void)in_sizes; (void)n_in; (void)out_size; (void)ws_size;
    const float* qkv   = (const float*)d_in[0];   // [2048,1024]
    const float* w_in  = (const float*)d_in[1];   // [1024,3072]
    const float* b_in  = (const float*)d_in[2];   // [3072]
    const float* w_out = (const float*)d_in[3];   // [1024,1024]
    const float* b_out = (const float*)d_in[4];   // [1024]
    float* out = (float*)d_out;                   // [2048,1024]

    const int S = 2048, E = 1024, E3 = 3072;
    char* ws = (char*)d_ws;
    size_t off = 0;
    unsigned short* qkv_bf  = (unsigned short*)(ws + off); off += (size_t)S * E * 2;
    unsigned short* w_in_bf = (unsigned short*)(ws + off); off += (size_t)E * E3 * 2;
    unsigned short* w_out_bf= (unsigned short*)(ws + off); off += (size_t)E * E * 2;
    unsigned short* q_bf    = (unsigned short*)(ws + off); off += (size_t)S * E * 2;
    unsigned short* k_bf    = (unsigned short*)(ws + off); off += (size_t)S * E * 2;
    unsigned short* v_bf    = (unsigned short*)(ws + off); off += (size_t)S * E * 2;
    unsigned short* ctx_bf  = (unsigned short*)(ws + off); off += (size_t)S * E * 2;

    cvt_f32_bf16<<<1024, 256, 0, stream>>>((const float2*)qkv,   (unsigned int*)qkv_bf,  (S * E) / 2);
    cvt_f32_bf16<<<1024, 256, 0, stream>>>((const float2*)w_in,  (unsigned int*)w_in_bf, (E * E3) / 2);
    cvt_f32_bf16<<<1024, 256, 0, stream>>>((const float2*)w_out, (unsigned int*)w_out_bf,(E * E) / 2);

    // QKV projection: [2048,1024] x [1024,3072] + b_in -> q(*1/8), k, v (bf16)
    gemm_bf16_kernel<0><<<dim3(E3 / 128, S / 64), 256, 0, stream>>>(
        qkv_bf, w_in_bf, b_in, q_bf, k_bf, v_bf, nullptr, S, E3, E);

    // fused attention: 16 heads x 16 query blocks of 128 rows
    attn_fwd_kernel<<<dim3(16, S / 128), 256, 0, stream>>>(q_bf, k_bf, v_bf, ctx_bf);

    // output projection: [2048,1024] x [1024,1024] + b_out -> fp32 out
    gemm_bf16_kernel<1><<<dim3(E / 128, S / 64), 256, 0, stream>>>(
        ctx_bf, w_out_bf, b_out, nullptr, nullptr, nullptr, out, S, E, E);
}
